// AaD_57956288692799
// MI455X (gfx1250) — compile-verified, matrix operand order checked by reading, not run on hardware
//
#include <hip/hip_runtime.h>
#include <hip/hip_bf16.h>
#include <math.h>

#define B_SZ 256
#define D_SZ 512
#define C_SZ 345
#define N_SZ 50000
#define K_NEAR 5
#define ALPHA_C 1.0f
#define EPS_C 1e-12f
#define COL_GROUPS 2   // 128-column groups per block

typedef __attribute__((ext_vector_type(16))) __bf16 v16bf;
typedef __attribute__((ext_vector_type(8)))  float  v8f;

__device__ __forceinline__ __bf16 cvt_bf16(float x) { return (__bf16)x; }

// ---------------- softmax rows + per-row sum of squares ----------------
__global__ void softmax_kernel(const float* __restrict__ pred,
                               float* __restrict__ S,
                               float* __restrict__ rowsumsq) {
  int b = blockIdx.x, tid = threadIdx.x; // 128 threads
  __shared__ float red[128];
  const float* row = pred + b * C_SZ;
  float m = -INFINITY;
  for (int c = tid; c < C_SZ; c += 128) m = fmaxf(m, row[c]);
  red[tid] = m; __syncthreads();
  for (int s = 64; s > 0; s >>= 1) { if (tid < s) red[tid] = fmaxf(red[tid], red[tid + s]); __syncthreads(); }
  m = red[0]; __syncthreads();
  float sum = 0.f;
  for (int c = tid; c < C_SZ; c += 128) { float e = expf(row[c] - m); S[b * C_SZ + c] = e; sum += e; }
  red[tid] = sum; __syncthreads();
  for (int s = 64; s > 0; s >>= 1) { if (tid < s) red[tid] += red[tid + s]; __syncthreads(); }
  float inv = 1.0f / red[0]; __syncthreads();
  float ss = 0.f;
  for (int c = tid; c < C_SZ; c += 128) { float p = S[b * C_SZ + c] * inv; S[b * C_SZ + c] = p; ss += p * p; }
  red[tid] = ss; __syncthreads();
  for (int s = 64; s > 0; s >>= 1) { if (tid < s) red[tid] += red[tid + s]; __syncthreads(); }
  if (tid == 0) rowsumsq[b] = red[0];
}

// ---------------- L2-normalize feature rows ----------------
__global__ void norm_kernel(const float* __restrict__ feat, float* __restrict__ F) {
  int b = blockIdx.x, tid = threadIdx.x; // 128 threads
  __shared__ float red[128];
  float ss = 0.f;
  for (int d = tid; d < D_SZ; d += 128) { float v = feat[b * D_SZ + d]; ss += v * v; }
  red[tid] = ss; __syncthreads();
  for (int s = 64; s > 0; s >>= 1) { if (tid < s) red[tid] += red[tid + s]; __syncthreads(); }
  float inv = 1.0f / fmaxf(sqrtf(red[0]), EPS_C);
  for (int d = tid; d < D_SZ; d += 128) F[b * D_SZ + d] = feat[b * D_SZ + d] * inv;
}

// ---------------- override map: bank row -> batch row (last occurrence) ----------------
__global__ void override_init(int* __restrict__ ovr) {
  int n = blockIdx.x * 256 + threadIdx.x;
  if (n < N_SZ) ovr[n] = -1;
}
__global__ void override_set(const int* __restrict__ trg, int* __restrict__ ovr) {
  int r = threadIdx.x; // 256 threads, 1 block
  int n = trg[r];
  bool last = true;
  for (int r2 = r + 1; r2 < B_SZ; ++r2) if (trg[r2] == n) { last = false; break; }
  if (last) ovr[n] = r;
}

// ---------------- WMMA GEMM: out[256 x ncols] = F(256x512) * Bmat(ncols x 512)^T ----------------
// A packed once per block into LDS as bf16 in WMMA A-fragment order (256 KB of 320 KB WGP LDS).
// Each block sweeps COL_GROUPS column groups of 128 (16 cols/wave). Per group each wave keeps all
// 16 K-step B fragments resident in VGPRs and runs row tiles in pairs (2 independent WMMA chains).
// A fragments are double-buffered from LDS; sched_barrier(0) pins the prefetch ahead of the WMMA
// pair so the ds_load latency hides behind WMMA execution (wait becomes dscnt<=4, not 0).
__global__ __launch_bounds__(256) void dist_gemm(const float* __restrict__ F,
                                                 const float* __restrict__ Bmat,
                                                 float* __restrict__ out,
                                                 int ncols, int ld) {
  __shared__ __bf16 Alds[B_SZ * D_SZ]; // 131072 bf16 = 256 KB
  int tid = threadIdx.x;

  // Pack A: fragment f = m*16+s laid out as [f][lane(32)][16 bf16] (32B contiguous per lane).
  for (int e = 0; e < 128; ++e) {
    int q   = e * 256 + tid;          // quad index (coalesced global reads)
    int row = q >> 7;                 // 128 quads per row
    int kq  = (q & 127) << 2;         // k, multiple of 4 -> stays inside one 8-group
    float4 f4 = *reinterpret_cast<const float4*>(F + row * D_SZ + kq);
    int m = row >> 4, lm = row & 15;
    int s = kq >> 5, ko = kq & 31, grp = ko >> 3;
    int lane = (grp & 1) ? (lm + 16) : lm;          // A layout: lanes<16 hold K{0..7,16..23}
    int idx  = (grp < 2) ? (ko & 7) : ((ko & 7) + 8);
    int off  = ((m * 16 + s) * 32 + lane) * 16 + idx;
    union { __bf16 h[4]; uint2 u; } pk;
    pk.h[0] = cvt_bf16(f4.x); pk.h[1] = cvt_bf16(f4.y);
    pk.h[2] = cvt_bf16(f4.z); pk.h[3] = cvt_bf16(f4.w);
    *reinterpret_cast<uint2*>(&Alds[off]) = pk.u;   // 8B aligned (idx % 4 == 0)
  }
  __syncthreads();

  int wave = tid >> 5, lane = tid & 31;
  int cl   = lane & 15;
  int koff = (lane & 16) ? 8 : 0;                   // B layout mirrors A (lanes<16: K{0..7,16..23})
  const __bf16* alane = Alds + lane * 16;           // lane-local base into fragment array

  for (int g = 0; g < COL_GROUPS; ++g) {
    int colbase = (blockIdx.x * COL_GROUPS + g) * 128 + wave * 16;
    if (colbase >= ncols) continue;                 // wave-uniform: EXEC all-ones for WMMA
    int col  = colbase + cl;
    int colc = (col < ncols) ? col : (ncols - 1);   // clamp OOB loads, mask on store
    const float* bp = Bmat + (size_t)colc * D_SZ;

    v16bf bfrag[16];
#pragma unroll
    for (int s = 0; s < 16; ++s) {
      const float* p = bp + s * 32 + koff;
      float4 x0 = *reinterpret_cast<const float4*>(p);
      float4 x1 = *reinterpret_cast<const float4*>(p + 4);
      float4 x2 = *reinterpret_cast<const float4*>(p + 16);
      float4 x3 = *reinterpret_cast<const float4*>(p + 20);
      v16bf bb;
      bb[0]  = cvt_bf16(x0.x); bb[1]  = cvt_bf16(x0.y); bb[2]  = cvt_bf16(x0.z); bb[3]  = cvt_bf16(x0.w);
      bb[4]  = cvt_bf16(x1.x); bb[5]  = cvt_bf16(x1.y); bb[6]  = cvt_bf16(x1.z); bb[7]  = cvt_bf16(x1.w);
      bb[8]  = cvt_bf16(x2.x); bb[9]  = cvt_bf16(x2.y); bb[10] = cvt_bf16(x2.z); bb[11] = cvt_bf16(x2.w);
      bb[12] = cvt_bf16(x3.x); bb[13] = cvt_bf16(x3.y); bb[14] = cvt_bf16(x3.z); bb[15] = cvt_bf16(x3.w);
      bfrag[s] = bb;
    }

    for (int m = 0; m < 16; m += 2) {               // 2 row tiles -> 2 independent WMMA chains
      const __bf16* base0 = alane + ((m * 16) * 32) * 16;
      const __bf16* base1 = alane + (((m + 1) * 16) * 32) * 16;
      v8f acc0 = {0.f, 0.f, 0.f, 0.f, 0.f, 0.f, 0.f, 0.f};
      v8f acc1 = {0.f, 0.f, 0.f, 0.f, 0.f, 0.f, 0.f, 0.f};
      v16bf a0 = *reinterpret_cast<const v16bf*>(base0);
      v16bf a1 = *reinterpret_cast<const v16bf*>(base1);
#pragma unroll
      for (int s = 0; s < 16; ++s) {
        v16bf a0n, a1n;
        if (s < 15) {                               // prefetch next K-step fragments
          a0n = *reinterpret_cast<const v16bf*>(base0 + (s + 1) * 512);
          a1n = *reinterpret_cast<const v16bf*>(base1 + (s + 1) * 512);
        } else {
          a0n = a0; a1n = a1;
        }
        // Pin the prefetch: nothing may move across this point, so the ds_loads above
        // stay ahead of the WMMA pair and the wait before the WMMAs is dscnt<=4, not 0.
        __builtin_amdgcn_sched_barrier(0);
        acc0 = __builtin_amdgcn_wmma_f32_16x16x32_bf16(false, a0, false, bfrag[s],
                                                       (short)0, acc0, false, false);
        acc1 = __builtin_amdgcn_wmma_f32_16x16x32_bf16(false, a1, false, bfrag[s],
                                                       (short)0, acc1, false, false);
        __builtin_amdgcn_sched_barrier(0);
        a0 = a0n; a1 = a1n;
      }
      int rbase = m * 16 + ((lane & 16) ? 8 : 0);   // C/D layout: VGPR j -> M=j (lanes<16), M=j+8
      if (col < ncols) {
#pragma unroll
        for (int j = 0; j < 8; ++j) {
          out[(size_t)(rbase + j) * ld + col]      = acc0[j];
          out[(size_t)(rbase + 16 + j) * ld + col] = acc1[j];
        }
      }
    }
  }
}

// ---------------- fixup of overridden columns: dist[:, trg[r]] = G[:, r] ----------------
__global__ void fixup_kernel(const int* __restrict__ trg, const int* __restrict__ ovr,
                             const float* __restrict__ G, float* __restrict__ dist) {
  int r = blockIdx.x, i = threadIdx.x;
  int n = trg[r];
  if (ovr[n] != r) return;                // only last occurrence wins (block-uniform)
  dist[(size_t)i * N_SZ + n] = G[i * B_SZ + r];
}

// ---------------- per-row top-6 (desc, ties -> lower index), keep entries 1..5 ----------------
__global__ __launch_bounds__(256) void topk_kernel(const float* __restrict__ dist,
                                                   int* __restrict__ idx_near) {
  int b = blockIdx.x, tid = threadIdx.x;
  const float* row = dist + (size_t)b * N_SZ;
  float tv[6]; int ti[6];
#pragma unroll
  for (int k = 0; k < 6; ++k) { tv[k] = -INFINITY; ti[k] = 0x7fffffff; }
  for (int n = tid; n < N_SZ; n += 256) {
    float v = row[n];
    if (v > tv[5] || (v == tv[5] && n < ti[5])) {
      int p = 5;
      while (p > 0 && (v > tv[p - 1] || (v == tv[p - 1] && n < ti[p - 1]))) {
        tv[p] = tv[p - 1]; ti[p] = ti[p - 1]; --p;
      }
      tv[p] = v; ti[p] = n;
    }
  }
  __shared__ float sv[256 * 6];
  __shared__ int   si[256 * 6];
#pragma unroll
  for (int k = 0; k < 6; ++k) { sv[tid * 6 + k] = tv[k]; si[tid * 6 + k] = ti[k]; }
  __syncthreads();
  if (tid == 0) {                          // deterministic 256-way merge
    int head[256];
    for (int t = 0; t < 256; ++t) head[t] = 0;
    for (int k = 0; k < 6; ++k) {
      float bv = -INFINITY; int bi = 0x7fffffff; int bt = 0;
      for (int t = 0; t < 256; ++t) {
        int h = head[t];
        if (h < 6) {
          float v = sv[t * 6 + h]; int i2 = si[t * 6 + h];
          if (v > bv || (v == bv && i2 < bi)) { bv = v; bi = i2; bt = t; }
        }
      }
      head[bt]++;
      if (k >= 1) idx_near[b * K_NEAR + (k - 1)] = bi;   // drop argmax
    }
  }
}

// ---------------- KL term per batch row: sum_k sum_c s*(log s - p_b) ----------------
__global__ void kl_kernel(const float* __restrict__ S, const float* __restrict__ SB,
                          const int* __restrict__ idxn, const int* __restrict__ ovr,
                          float* __restrict__ klpart) {
  int b = blockIdx.x, tid = threadIdx.x; // 256 threads
  __shared__ float red[256];
  float acc = 0.f;
  for (int t = tid; t < K_NEAR * C_SZ; t += 256) {
    int k = t / C_SZ, c = t - k * C_SZ;
    int n = idxn[b * K_NEAR + k];
    int r = ovr[n];
    float s = (r >= 0) ? S[r * C_SZ + c] : SB[(size_t)n * C_SZ + c];
    acc += s * (logf(s) - S[b * C_SZ + c]);
  }
  red[tid] = acc; __syncthreads();
  for (int st = 128; st > 0; st >>= 1) { if (tid < st) red[tid] += red[tid + st]; __syncthreads(); }
  if (tid == 0) klpart[b] = red[0];
}

// ---------------- column sums of S (for neg_pred closed form) ----------------
__global__ void colsum_kernel(const float* __restrict__ S, float* __restrict__ colsum) {
  int c = threadIdx.x; // 384 threads, 1 block
  if (c < C_SZ) {
    float s = 0.f;
    for (int b = 0; b < B_SZ; ++b) s += S[b * C_SZ + c];
    colsum[c] = s;
  }
}

// ---------------- final scalar: kl/B + (||colsum||^2 - sum rowsumsq)/B ----------------
__global__ void final_kernel(const float* __restrict__ klpart, const float* __restrict__ colsum,
                             const float* __restrict__ rowsumsq, float* __restrict__ out) {
  if (threadIdx.x == 0 && blockIdx.x == 0) {
    float kls = 0.f; for (int b = 0; b < B_SZ; ++b) kls += klpart[b];
    float cn  = 0.f; for (int c = 0; c < C_SZ; ++c) cn += colsum[c] * colsum[c];
    float rs  = 0.f; for (int b = 0; b < B_SZ; ++b) rs += rowsumsq[b];
    out[0] = kls / (float)B_SZ + ALPHA_C * ((cn - rs) / (float)B_SZ);
  }
}

extern "C" void kernel_launch(void* const* d_in, const int* in_sizes, int n_in,
                              void* d_out, int out_size, void* d_ws, size_t ws_size,
                              hipStream_t stream) {
  (void)in_sizes; (void)n_in; (void)out_size; (void)ws_size;
  const float* features    = (const float*)d_in[0];
  const float* predictions = (const float*)d_in[1];
  const float* fea_bank    = (const float*)d_in[2];
  const float* score_bank  = (const float*)d_in[3];
  const int*   trg_idx     = (const int*)d_in[4];   // harness passes integer inputs as int32
  float* out = (float*)d_out;

  char* ws = (char*)d_ws;
  size_t off = 0;
  auto alloc = [&](size_t bytes) -> void* {
    void* p = ws + off;
    off += (bytes + 255) & ~(size_t)255;
    return p;
  };
  float* S        = (float*)alloc((size_t)B_SZ * C_SZ * 4);
  float* F        = (float*)alloc((size_t)B_SZ * D_SZ * 4);
  float* G        = (float*)alloc((size_t)B_SZ * B_SZ * 4);
  float* dist     = (float*)alloc((size_t)B_SZ * N_SZ * 4);   // 51.2 MB
  int*   ovr      = (int*)alloc((size_t)N_SZ * 4);
  int*   idxn     = (int*)alloc((size_t)B_SZ * K_NEAR * 4);
  float* klpart   = (float*)alloc((size_t)B_SZ * 4);
  float* rowsumsq = (float*)alloc((size_t)B_SZ * 4);
  float* colsum   = (float*)alloc((size_t)C_SZ * 4);

  const int colsPerBlock = COL_GROUPS * 128;

  softmax_kernel<<<B_SZ, 128, 0, stream>>>(predictions, S, rowsumsq);
  norm_kernel<<<B_SZ, 128, 0, stream>>>(features, F);
  override_init<<<(N_SZ + 255) / 256, 256, 0, stream>>>(ovr);
  override_set<<<1, B_SZ, 0, stream>>>(trg_idx, ovr);
  dist_gemm<<<(B_SZ + colsPerBlock - 1) / colsPerBlock, 256, 0, stream>>>(F, F, G, B_SZ, B_SZ);
  dist_gemm<<<(N_SZ + colsPerBlock - 1) / colsPerBlock, 256, 0, stream>>>(F, fea_bank, dist, N_SZ, N_SZ);
  fixup_kernel<<<B_SZ, B_SZ, 0, stream>>>(trg_idx, ovr, G, dist);
  topk_kernel<<<B_SZ, 256, 0, stream>>>(dist, idxn);
  kl_kernel<<<B_SZ, 256, 0, stream>>>(S, score_bank, idxn, ovr, klpart);
  colsum_kernel<<<1, 384, 0, stream>>>(S, colsum);
  final_kernel<<<1, 64, 0, stream>>>(klpart, colsum, rowsumsq, out);
}
